// BertSelfAttention_17471926960662
// MI455X (gfx1250) — matrix operand order repaired and checked by reading, not compiled, not run on hardware
//
#include <hip/hip_runtime.h>
#include <hip/hip_bf16.h>

typedef __attribute__((ext_vector_type(16))) __bf16 v16bf;
typedef __attribute__((ext_vector_type(8)))  float  v8f;
typedef __attribute__((ext_vector_type(4)))  float          f32x4;
typedef __attribute__((ext_vector_type(4)))  unsigned short u16x4;

union FragBF { v16bf v; uint4 q[2]; };

static __device__ inline unsigned short f32_to_bf16(float f) {
  union { float f; unsigned u; } x; x.f = f;
  unsigned r = x.u + 0x7FFFu + ((x.u >> 16) & 1u);   // round-to-nearest-even
  return (unsigned short)(r >> 16);
}

static __device__ inline v8f zero_v8f() {
  v8f z = {0.f,0.f,0.f,0.f,0.f,0.f,0.f,0.f};
  return z;
}

// ---------------------------------------------------------------------------
// fp32 -> bf16 convert (streaming; nontemporal)
// ---------------------------------------------------------------------------
__global__ __launch_bounds__(256)
void cvt_f32_bf16_kernel(const float* __restrict__ src,
                         unsigned short* __restrict__ dst, int n) {
  int i = (blockIdx.x * blockDim.x + threadIdx.x) * 4;
  if (i + 3 < n) {
    f32x4 f = __builtin_nontemporal_load((const f32x4*)(src + i));
    u16x4 o;
    o.x = f32_to_bf16(f.x); o.y = f32_to_bf16(f.y);
    o.z = f32_to_bf16(f.z); o.w = f32_to_bf16(f.w);
    __builtin_nontemporal_store(o, (u16x4*)(dst + i));
  } else {
    for (int j = i; j < n; ++j) dst[j] = f32_to_bf16(src[j]);
  }
}

// ---------------------------------------------------------------------------
// bf16 WMMA GEMM:  C[M,N] = A[M,K] @ W[N,K]^T + bias   (M=4096, K=N=1024)
// Block tile 128x128, 8 waves (2 M x 4 N), wave tile 64x32 (4x2 WMMA tiles).
// Software-pipelined: next K-tile loaded to VGPRs while WMMAs run.
// MODE 0: fp32 row-major out.  MODE 1: bf16 out in [B=2,H=16,S=2048,64] layout.
// ---------------------------------------------------------------------------
template<int MODE>
__global__ __launch_bounds__(256)
void gemm_bf16_kernel(const unsigned short* __restrict__ A,
                      const unsigned short* __restrict__ Bw,
                      const float* __restrict__ bias,
                      void* __restrict__ out, float scale) {
  constexpr int Kt = 1024, Nt = 1024;
  constexpr int BM = 128, BN = 128, BK = 32;
  constexpr int LDA = BK + 8;                      // 40 ushorts = 80B rows
  __shared__ alignas(16) unsigned short As[BM * LDA];
  __shared__ alignas(16) unsigned short Bs[BN * LDA];

  const int t    = threadIdx.x;
  const int wave = t >> 5, lane = t & 31;
  const int wm   = wave & 1, wn = wave >> 1;       // 2 x 4 wave grid
  const int half = lane >> 4, l16 = lane & 15;
  const int bm0  = (int)blockIdx.x * BM;
  const int bn0  = (int)blockIdx.y * BN;
  const int klo  = half ? 8 : 0;

  const int r0 = t >> 2;                           // staging row (0..63)
  const int g0 = (t & 3) * 8;                      // staging col group

  v8f acc[4][2];
  #pragma unroll
  for (int i = 0; i < 4; ++i)
    #pragma unroll
    for (int j = 0; j < 2; ++j) acc[i][j] = zero_v8f();

  uint4 ra0, ra1, rb0, rb1;
  {
    ra0 = *(const uint4*)(A  + (size_t)(bm0 + r0)      * Kt + g0);
    ra1 = *(const uint4*)(A  + (size_t)(bm0 + 64 + r0) * Kt + g0);
    rb0 = *(const uint4*)(Bw + (size_t)(bn0 + r0)      * Kt + g0);
    rb1 = *(const uint4*)(Bw + (size_t)(bn0 + 64 + r0) * Kt + g0);
  }

  for (int kk = 0; kk < Kt; kk += BK) {
    *(uint4*)(As + r0 * LDA + g0)        = ra0;
    *(uint4*)(As + (64 + r0) * LDA + g0) = ra1;
    *(uint4*)(Bs + r0 * LDA + g0)        = rb0;
    *(uint4*)(Bs + (64 + r0) * LDA + g0) = rb1;
    __syncthreads();

    if (kk + BK < Kt) {                            // prefetch next tile to regs
      int kn = kk + BK;
      ra0 = *(const uint4*)(A  + (size_t)(bm0 + r0)      * Kt + kn + g0);
      ra1 = *(const uint4*)(A  + (size_t)(bm0 + 64 + r0) * Kt + kn + g0);
      rb0 = *(const uint4*)(Bw + (size_t)(bn0 + r0)      * Kt + kn + g0);
      rb1 = *(const uint4*)(Bw + (size_t)(bn0 + 64 + r0) * Kt + kn + g0);
    }

    FragBF a[4], b[2];
    #pragma unroll
    for (int i = 0; i < 4; ++i) {                  // A frags (16x32 bf16)
      const unsigned short* src = As + (wm * 64 + i * 16 + l16) * LDA + klo;
      a[i].q[0] = *(const uint4*)src;
      a[i].q[1] = *(const uint4*)(src + 16);
    }
    #pragma unroll
    for (int j = 0; j < 2; ++j) {                  // B frags (32x16 bf16)
      const unsigned short* src = Bs + (wn * 32 + j * 16 + l16) * LDA + klo;
      b[j].q[0] = *(const uint4*)src;
      b[j].q[1] = *(const uint4*)(src + 16);
    }
    #pragma unroll
    for (int i = 0; i < 4; ++i)
      #pragma unroll
      for (int j = 0; j < 2; ++j)
        acc[i][j] = __builtin_amdgcn_wmma_f32_16x16x32_bf16(
            false, a[i].v, false, b[j].v, (short)0, acc[i][j], false, false);
    __syncthreads();
  }

  const int gm_base = bm0 + wm * 64;
  const int gn_base = bn0 + wn * 32;
  #pragma unroll
  for (int i = 0; i < 4; ++i) {
    #pragma unroll
    for (int j = 0; j < 2; ++j) {
      int gn = gn_base + j * 16 + l16;
      float bv = bias[gn];
      #pragma unroll
      for (int v = 0; v < 8; ++v) {
        int gm = gm_base + i * 16 + v + 8 * half;
        float val = (acc[i][j][v] + bv) * scale;
        if (MODE == 0) {
          __builtin_nontemporal_store(val, (float*)out + (size_t)gm * Nt + gn);
        } else {
          int b_ = gm >> 11, s_ = gm & 2047;       // [B*S] -> (b,s)
          int h_ = gn >> 6,  d_ = gn & 63;         // [H*64] -> (h,d)
          __builtin_nontemporal_store(f32_to_bf16(val),
              (unsigned short*)out + (((size_t)(b_ * 16 + h_) * 2048) + s_) * 64 + d_);
        }
      }
    }
  }
}

// ---------------------------------------------------------------------------
// Flash attention with identity mask.  Q pre-scaled by 1/8.
// Q,K,V: bf16 [B,H,S,64].  Ctx: bf16 [B,S,1024].
// Block: 128 threads (4 waves), each wave owns 32 q-rows (2 q-tiles);
// 128 q-rows per block, keys staged in LDS in chunks of 32.
// Row-sum of P via WMMA against an all-ones B matrix; diagonal mask applied
// under a wave-uniform branch (diagonal hits a q-tile in only 1/64 chunks).
// ---------------------------------------------------------------------------
__global__ __launch_bounds__(128)
void attn_kernel(const unsigned short* __restrict__ Q,
                 const unsigned short* __restrict__ K,
                 const unsigned short* __restrict__ V,
                 unsigned short* __restrict__ Ctx) {
  constexpr int S = 2048, DH = 64, H = 16;
  constexpr int LK = DH + 8;                       // Ks row stride (72)
  constexpr int LV = 32 + 8;                       // Vt row stride (40, 80B rows)
  __shared__ alignas(16) unsigned short Ks[32 * LK];
  __shared__ alignas(16) unsigned short Vt[DH * LV];
  __shared__ alignas(16) unsigned short Ps[4 * 2 * 16 * 32];

  const int t    = threadIdx.x;
  const int wave = t >> 5, lane = t & 31;
  const int half = lane >> 4, l16 = lane & 15;
  const int qblk = blockIdx.x & 15;                // S/128 = 16 q-blocks
  const int h    = (blockIdx.x >> 4) & 15;
  const int b    = blockIdx.x >> 8;
  const int qbw  = qblk * 128 + wave * 32;         // wave's first q-row
  const int klo  = half ? 8 : 0;

  const unsigned short* Qh = Q + (size_t)(b * H + h) * S * DH;
  const unsigned short* Kh = K + (size_t)(b * H + h) * S * DH;
  const unsigned short* Vh = V + (size_t)(b * H + h) * S * DH;

  FragBF aq[2][2];                                 // Q frags live in regs
  #pragma unroll
  for (int qt = 0; qt < 2; ++qt)
    #pragma unroll
    for (int i = 0; i < 2; ++i) {
      const unsigned short* src =
          Qh + (size_t)(qbw + qt * 16 + l16) * DH + i * 32 + klo;
      aq[qt][i].q[0] = *(const uint4*)src;
      aq[qt][i].q[1] = *(const uint4*)(src + 16);
    }

  FragBF fone;                                     // all-ones bf16 B matrix
  fone.q[0] = make_uint4(0x3F803F80u, 0x3F803F80u, 0x3F803F80u, 0x3F803F80u);
  fone.q[1] = fone.q[0];

  v8f O[2][4];
  #pragma unroll
  for (int qt = 0; qt < 2; ++qt)
    #pragma unroll
    for (int dt = 0; dt < 4; ++dt) O[qt][dt] = zero_v8f();
  float mrow[2][8], lrow[2][8];
  #pragma unroll
  for (int qt = 0; qt < 2; ++qt)
    #pragma unroll
    for (int v = 0; v < 8; ++v) { mrow[qt][v] = -3.0e38f; lrow[qt][v] = 0.f; }

  for (int kb = 0; kb < S; kb += 32) {
    __syncthreads();
    #pragma unroll
    for (int p = 0; p < 2; ++p) {                  // stage K chunk 32x64
      int lin = p * 128 + t;
      int row = lin >> 3, g = lin & 7;
      uint4 d = *(const uint4*)(Kh + (size_t)(kb + row) * DH + g * 8);
      *(uint4*)(Ks + row * LK + g * 8) = d;
    }
    {                                              // stage V transposed, packed
      int kp = t >> 3;                             // key pair 0..15
      int g  = (t & 7) * 8;                        // dim group
      uint4 va = *(const uint4*)(Vh + (size_t)(kb + 2 * kp)     * DH + g);
      uint4 vb = *(const uint4*)(Vh + (size_t)(kb + 2 * kp + 1) * DH + g);
      const unsigned short* ha = (const unsigned short*)&va;
      const unsigned short* hb = (const unsigned short*)&vb;
      #pragma unroll
      for (int j = 0; j < 8; ++j) {
        unsigned pk = (unsigned)ha[j] | ((unsigned)hb[j] << 16);
        *(unsigned*)(Vt + (g + j) * LV + 2 * kp) = pk;
      }
    }
    if (kb + 32 < S) {                             // prefetch next chunk
      __builtin_prefetch(Kh + (size_t)(kb + 32) * DH + t * 16, 0, 0);
      __builtin_prefetch(Vh + (size_t)(kb + 32) * DH + t * 16, 0, 0);
    }
    __syncthreads();

    FragBF bk[2][2];                               // all K frags up-front
    #pragma unroll
    for (int tt = 0; tt < 2; ++tt)
      #pragma unroll
      for (int i = 0; i < 2; ++i) {
        const unsigned short* src =
            Ks + (tt * 16 + l16) * LK + i * 32 + klo;
        bk[tt][i].q[0] = *(const uint4*)src;
        bk[tt][i].q[1] = *(const uint4*)(src + 16);
      }

    v8f sc[2][2];                                  // scores: 2 q-tiles x 2 k-tiles
    #pragma unroll
    for (int qt = 0; qt < 2; ++qt)
      #pragma unroll
      for (int tt = 0; tt < 2; ++tt) {
        v8f s = __builtin_amdgcn_wmma_f32_16x16x32_bf16(
            false, aq[qt][0].v, false, bk[tt][0].v, (short)0, zero_v8f(), false, false);
        sc[qt][tt] = __builtin_amdgcn_wmma_f32_16x16x32_bf16(
            false, aq[qt][1].v, false, bk[tt][1].v, (short)0, s, false, false);
      }

    #pragma unroll
    for (int qt = 0; qt < 2; ++qt) {               // identity mask
      int qb = qbw + qt * 16;
      if (kb < qb + 16 && qb < kb + 32) {          // wave-uniform branch
        #pragma unroll
        for (int tt = 0; tt < 2; ++tt) {
          int kcol = kb + tt * 16 + l16;
          #pragma unroll
          for (int v = 0; v < 8; ++v)
            if (qb + v + 8 * half == kcol) sc[qt][tt][v] = -3.0e38f;
        }
      }
    }

    float alpha[2][8];                             // online softmax (max only)
    #pragma unroll
    for (int qt = 0; qt < 2; ++qt)
      #pragma unroll
      for (int v = 0; v < 8; ++v) {
        float mx = fmaxf(sc[qt][0][v], sc[qt][1][v]);
        #pragma unroll
        for (int off = 1; off < 16; off <<= 1)
          mx = fmaxf(mx, __shfl_xor(mx, off, 32));
        float mnew = fmaxf(mrow[qt][v], mx);
        alpha[qt][v] = __expf(mrow[qt][v] - mnew);
        sc[qt][0][v] = __expf(sc[qt][0][v] - mnew);
        sc[qt][1][v] = __expf(sc[qt][1][v] - mnew);
        mrow[qt][v] = mnew;
      }
    #pragma unroll
    for (int qt = 0; qt < 2; ++qt)                 // rescale O
      #pragma unroll
      for (int dt = 0; dt < 4; ++dt)
        #pragma unroll
        for (int v = 0; v < 8; ++v) O[qt][dt][v] *= alpha[qt][v];

    #pragma unroll
    for (int qt = 0; qt < 2; ++qt) {               // P -> LDS as bf16
      unsigned short* Pw = Ps + (wave * 2 + qt) * 512;
      #pragma unroll
      for (int tt = 0; tt < 2; ++tt)
        #pragma unroll
        for (int v = 0; v < 8; ++v)
          Pw[(v + 8 * half) * 32 + tt * 16 + l16] = f32_to_bf16(sc[qt][tt][v]);
    }
    __syncthreads();

    FragBF bv[4];                                  // all V frags up-front
    #pragma unroll
    for (int dt = 0; dt < 4; ++dt) {
      const unsigned short* src = Vt + (dt * 16 + l16) * LV + klo;
      bv[dt].q[0] = *(const uint4*)src;
      bv[dt].q[1] = *(const uint4*)(src + 16);
    }
    FragBF ap[2];                                  // re-fragment P (A layout)
    #pragma unroll
    for (int qt = 0; qt < 2; ++qt) {
      const unsigned short* src = Ps + (wave * 2 + qt) * 512 + l16 * 32 + klo;
      ap[qt].q[0] = *(const uint4*)src;
      ap[qt].q[1] = *(const uint4*)(src + 16);
    }

    #pragma unroll
    for (int qt = 0; qt < 2; ++qt) {
      // row-sum via matrix pipe: P @ ones -> every col = row sum
      v8f ls = __builtin_amdgcn_wmma_f32_16x16x32_bf16(
          false, ap[qt].v, false, fone.v, (short)0, zero_v8f(), false, false);
      #pragma unroll
      for (int v = 0; v < 8; ++v)
        lrow[qt][v] = lrow[qt][v] * alpha[qt][v] + ls[v];
      #pragma unroll
      for (int dt = 0; dt < 4; ++dt)               // O += P @ V
        O[qt][dt] = __builtin_amdgcn_wmma_f32_16x16x32_bf16(
            false, ap[qt].v, false, bv[dt].v, (short)0, O[qt][dt], false, false);
    }
  }

  #pragma unroll
  for (int qt = 0; qt < 2; ++qt)                   // normalize + store ctx
    #pragma unroll
    for (int dt = 0; dt < 4; ++dt)
      #pragma unroll
      for (int v = 0; v < 8; ++v) {
        int qrow = qbw + qt * 16 + v + 8 * half;
        float val = O[qt][dt][v] / lrow[qt][v];
        __builtin_nontemporal_store(f32_to_bf16(val),
            Ctx + (size_t)(b * S + qrow) * 1024 + h * DH + dt * 16 + l16);
      }
}

// ---------------------------------------------------------------------------
extern "C" void kernel_launch(void* const* d_in, const int* in_sizes, int n_in,
                              void* d_out, int out_size, void* d_ws, size_t ws_size,
                              hipStream_t stream) {
  (void)in_sizes; (void)n_in; (void)out_size; (void)ws_size;
  const float* hidden = (const float*)d_in[0];
  const float* Wq = (const float*)d_in[1];
  const float* bq = (const float*)d_in[2];
  const float* Wk = (const float*)d_in[3];
  const float* bk = (const float*)d_in[4];
  const float* Wv = (const float*)d_in[5];
  const float* bv = (const float*)d_in[6];
  const float* Wd = (const float*)d_in[7];
  const float* bd = (const float*)d_in[8];
  float* out = (float*)d_out;

  const size_t X_ELEMS = (size_t)4096 * 1024;      // 4 Mi elems
  const size_t W_ELEMS = (size_t)1024 * 1024;      // 1 Mi elems
  unsigned short* Xbf   = (unsigned short*)d_ws;
  unsigned short* Wqbf  = Xbf   + X_ELEMS;
  unsigned short* Wkbf  = Wqbf  + W_ELEMS;
  unsigned short* Wvbf  = Wkbf  + W_ELEMS;
  unsigned short* Wdbf  = Wvbf  + W_ELEMS;
  unsigned short* Qbf   = Wdbf  + W_ELEMS;
  unsigned short* Kbf   = Qbf   + X_ELEMS;
  unsigned short* Vbf   = Kbf   + X_ELEMS;
  unsigned short* Ctxbf = Vbf   + X_ELEMS;

  // fp32 -> bf16 conversions
  cvt_f32_bf16_kernel<<<(int)(X_ELEMS / (256 * 4)), 256, 0, stream>>>(hidden, Xbf, (int)X_ELEMS);
  cvt_f32_bf16_kernel<<<(int)(W_ELEMS / (256 * 4)), 256, 0, stream>>>(Wq, Wqbf, (int)W_ELEMS);
  cvt_f32_bf16_kernel<<<(int)(W_ELEMS / (256 * 4)), 256, 0, stream>>>(Wk, Wkbf, (int)W_ELEMS);
  cvt_f32_bf16_kernel<<<(int)(W_ELEMS / (256 * 4)), 256, 0, stream>>>(Wv, Wvbf, (int)W_ELEMS);
  cvt_f32_bf16_kernel<<<(int)(W_ELEMS / (256 * 4)), 256, 0, stream>>>(Wd, Wdbf, (int)W_ELEMS);

  dim3 ggrid(32, 8);                               // M/128 x N/128
  // QKV projections (Q pre-scaled by 1/sqrt(64) = 0.125)
  gemm_bf16_kernel<1><<<ggrid, 256, 0, stream>>>(Xbf, Wqbf, bq, (void*)Qbf, 0.125f);
  gemm_bf16_kernel<1><<<ggrid, 256, 0, stream>>>(Xbf, Wkbf, bk, (void*)Kbf, 1.0f);
  gemm_bf16_kernel<1><<<ggrid, 256, 0, stream>>>(Xbf, Wvbf, bv, (void*)Vbf, 1.0f);

  // flash attention: B*H*(S/128) = 2*16*16 = 512 blocks
  attn_kernel<<<512, 128, 0, stream>>>(Qbf, Kbf, Vbf, Ctxbf);

  // output projection -> fp32
  gemm_bf16_kernel<0><<<ggrid, 256, 0, stream>>>(Ctxbf, Wdbf, bd, (void*)out, 1.0f);
}
